// MaskedAttention_62723702391328
// MI455X (gfx1250) — matrix operand order repaired and checked by reading, not compiled, not run on hardware
//
#include <hip/hip_runtime.h>
#include <hip/hip_bf16.h>

// ---------------------------------------------------------------------------
// MaskedAttention for MI455X (gfx1250, wave32, WMMA + TDM).
// Contractions via v_wmma_f32_16x16x32_bf16; fp32 -> bf16 via v_cvt_pk_bf16_f32;
// b_nd staged to LDS via tensor_load_to_lds; LDS tiles stored fragment-major so
// every WMMA operand half is a single ds_load_b128; global->reg staging is
// double-buffered so loads overlap the WMMA compute of the previous K-slice.
// ---------------------------------------------------------------------------

typedef __attribute__((ext_vector_type(16))) __bf16        v16bf;
typedef __attribute__((ext_vector_type(2)))  __bf16        v2bf;
typedef __attribute__((ext_vector_type(8)))  float         v8f;
typedef __attribute__((ext_vector_type(8)))  unsigned int  v8u;
typedef __attribute__((ext_vector_type(4)))  unsigned int  v4u;
typedef __attribute__((ext_vector_type(2)))  unsigned int  v2u;
typedef __attribute__((ext_vector_type(4)))  unsigned int  u32x4;
typedef __attribute__((ext_vector_type(8)))  int           i32x8;
typedef __attribute__((ext_vector_type(4)))  int           i32x4;

#define WMMA_BF16(a, b, c) \
  __builtin_amdgcn_wmma_f32_16x16x32_bf16(false, (a), false, (b), (short)0, (c), false, false)

// Hardware bf16 pair convert -> v_cvt_pk_bf16_f32.
__device__ __forceinline__ unsigned int pack2(float x, float y) {
  v2bf p;
  p[0] = (__bf16)x;
  p[1] = (__bf16)y;
  return __builtin_bit_cast(unsigned int, p);
}
__device__ __forceinline__ unsigned short f2bf(float x) {
  __bf16 b = (__bf16)x;
  return __builtin_bit_cast(unsigned short, b);
}

// Build a 16xbf16 WMMA operand from two 16B-aligned LDS quads (2x ds_load_b128).
__device__ __forceinline__ v16bf frag16(const unsigned int* base, int lo, int hi) {
  v4u l = *(const v4u*)(base + lo);
  v4u h = *(const v4u*)(base + hi);
  v8u t;
  t[0] = l[0]; t[1] = l[1]; t[2] = l[2]; t[3] = l[3];
  t[4] = h[0]; t[5] = h[1]; t[6] = h[2]; t[7] = h[3];
  return __builtin_bit_cast(v16bf, t);
}

static __device__ const v8f kZero8 = {0.f, 0.f, 0.f, 0.f, 0.f, 0.f, 0.f, 0.f};

// ---------------------------------------------------------------------------
// Generic GEMM: C[M,N] = A[M,K] @ W[K,N] (+ bias[N]) (+ resid[M,N]).
// 128x128x32 tiles, 256 threads = 8 waves (4 along M x 2 along N),
// each wave owns a 32x64 patch = 2x4 WMMA accumulators.
// LDS holds bf16 (k,k+1) pairs, fragment-major:
//   AsP[row][pair]  -> A operand = quads at {kpA, kpA+8}
//   BsPT[col][pair] -> B operand = quads at {kpB, kpB+8}
// Row pitch 20 dwords (80B) keeps every quad 16B-aligned.
// ---------------------------------------------------------------------------
struct StageRegs {
  float4 fa[4];          // A tile: 4 x (row, 2 K-pairs)
  float2 w0[4], w1[4];   // W tile: 4 x (rows k,k+1 at col pair)
};

__device__ __forceinline__ void stage_load(StageRegs& s,
                                           const float* __restrict__ A,
                                           const float* __restrict__ W,
                                           int bM, int bN, int kb, int N, int K,
                                           int tid) {
#pragma unroll
  for (int it = 0; it < 4; ++it) {           // A: 128 rows x 8 quads
    int idx = tid + it * 256;                // 0..1023
    int r = idx >> 3, pq = idx & 7;
    s.fa[it] = *(const float4*)(A + (size_t)(bM + r) * K + kb + pq * 4);
  }
#pragma unroll
  for (int it = 0; it < 4; ++it) {           // W: 16 pairs x 64 col-pairs
    int idx = tid + it * 256;                // 0..1023
    int p = idx >> 6, c2 = (idx & 63) * 2;
    int n = bN + c2;
    if (n + 1 < N) {                         // N is even in all uses
      s.w0[it] = *(const float2*)(W + (size_t)(kb + p * 2)     * N + n);
      s.w1[it] = *(const float2*)(W + (size_t)(kb + p * 2 + 1) * N + n);
    } else {
      s.w0[it] = make_float2(0.f, 0.f);
      s.w1[it] = make_float2(0.f, 0.f);
    }
  }
}

__device__ __forceinline__ void stage_store(const StageRegs& s,
                                            unsigned int (*AsP)[20],
                                            unsigned int (*BsPT)[20],
                                            int tid) {
#pragma unroll
  for (int it = 0; it < 4; ++it) {
    int idx = tid + it * 256;
    int r = idx >> 3, pq = idx & 7;
    v2u u;
    u[0] = pack2(s.fa[it].x, s.fa[it].y);
    u[1] = pack2(s.fa[it].z, s.fa[it].w);
    *(v2u*)&AsP[r][pq * 2] = u;              // 8B-aligned ds_store_b64
  }
#pragma unroll
  for (int it = 0; it < 4; ++it) {
    int idx = tid + it * 256;
    int p = idx >> 6, c2 = (idx & 63) * 2;
    BsPT[c2][p]     = pack2(s.w0[it].x, s.w1[it].x);  // vertical (k,k+1) pairs
    BsPT[c2 + 1][p] = pack2(s.w0[it].y, s.w1[it].y);
  }
}

__global__ __launch_bounds__(256)
void wmma_gemm_bf16(const float* __restrict__ A, const float* __restrict__ W,
                    const float* __restrict__ bias, const float* __restrict__ resid,
                    float* __restrict__ C, int M, int N, int K) {
  __shared__ unsigned int AsP[128][20];   // [row][K-pair 0..15] (pitch 80B)
  __shared__ unsigned int BsPT[128][20];  // [col][K-pair 0..15] (pitch 80B)

  const int tid  = threadIdx.x;
  const int lane = tid & 31;
  const int wave = tid >> 5;
  const int wm   = wave & 3;              // wave row (x32)
  const int wn   = wave >> 2;             // wave col (x64)

  const int nTiles = (N + 127) >> 7;
  const int bM = (int)(blockIdx.x / nTiles) * 128;
  const int bN = (int)(blockIdx.x % nTiles) * 128;

  v8f acc[2][4];
#pragma unroll
  for (int mt = 0; mt < 2; ++mt)
#pragma unroll
    for (int nt = 0; nt < 4; ++nt) acc[mt][nt] = kZero8;

  const int mrow  = lane & 15;
  const int kpA   = (lane & 16) ? 4 : 0;  // A fragment pair offset (K+8)
  const int kpB   = (lane & 16) ? 4 : 0;  // B fragment pair offset (K+8)
  const int ncol  = lane & 15;

  StageRegs sr;
  stage_load(sr, A, W, bM, bN, 0, N, K, tid);    // prologue: slice 0 in regs

  for (int kb = 0; kb < K; kb += 32) {
    stage_store(sr, AsP, BsPT, tid);             // regs -> LDS (bf16 pairs)
    __syncthreads();

    if (kb + 32 < K) {
      stage_load(sr, A, W, bM, bN, kb + 32, N, K, tid);  // overlap with WMMA
      __builtin_prefetch(A + (size_t)(bM + (tid >> 1)) * K + kb + 64, 0, 1);
    }

#pragma unroll
    for (int mt = 0; mt < 2; ++mt) {
      v16bf af = frag16(&AsP[wm * 32 + mt * 16 + mrow][0], kpA, kpA + 8);
#pragma unroll
      for (int nt = 0; nt < 4; ++nt) {
        v16bf bf = frag16(&BsPT[wn * 64 + nt * 16 + ncol][0], kpB, kpB + 8);
        acc[mt][nt] = WMMA_BF16(af, bf, acc[mt][nt]);
      }
    }
    __syncthreads();
  }

  // --- epilogue: C layout => lane = col, VGPR v = row (m = v + 8*(lane>=16)) ---
  const int mAdd = (lane & 16) ? 8 : 0;
#pragma unroll
  for (int nt = 0; nt < 4; ++nt) {
    int col = bN + wn * 64 + nt * 16 + ncol;
    if (col >= N) continue;
    float badd = bias ? bias[col] : 0.f;
#pragma unroll
    for (int mt = 0; mt < 2; ++mt) {
#pragma unroll
      for (int v = 0; v < 8; ++v) {
        int row = bM + wm * 32 + mt * 16 + v + mAdd;
        float val = acc[mt][nt][v] + badd;
        if (resid) val += resid[(size_t)row * N + col];
        C[(size_t)row * N + col] = val;
      }
    }
  }
}

// ---------------------------------------------------------------------------
// Attention: per block = (b, h, half). 4 waves x 16 q-rows = 64 rows/block.
// S = qk^T/8 + band bias + R.b_nd, softmax over T=256, then A = P @ V.
// P round-trips through LDS (bf16) to convert C-layout -> A-fragment layout.
// b_nd (10x128 f32, 5120B contiguous) is staged via the Tensor Data Mover.
// ---------------------------------------------------------------------------
__global__ __launch_bounds__(128)
void wmma_attention(const float* __restrict__ q,        // [B,128,1024]
                    const float* __restrict__ k_cache,  // [B,128,1024]
                    const float* __restrict__ v_cache,  // [B,128,1024]
                    const float* __restrict__ k_new,    // [B,128,1024]
                    const float* __restrict__ v_new,    // [B,128,1024]
                    const float* __restrict__ Rws,      // [B,128,160]
                    const float* __restrict__ b_nd,     // [10,128]
                    float* __restrict__ Aout) {         // [B,128,1024]
  const int NBASIS = 10, MAXL = 128, E = 1024, HDIM = 64, TQ = 128;

  __shared__ float sBnd[10][128];            // band basis (TDM destination)
  __shared__ unsigned short sP[4][16][256];  // per-wave P tiles (bf16)  32KB
  __shared__ float sR[64][12];               // R rows for this block

  const int tid  = threadIdx.x;
  const int lane = tid & 31;
  const int wave = tid >> 5;                 // 0..3
  const int blk  = blockIdx.x;
  const int half = blk & 1;
  const int h    = (blk >> 1) & 15;
  const int b    = blk >> 5;
  const int rBlk = half * 64;                // first q-row of block
  const int rW   = rBlk + wave * 16;         // first q-row of wave

  for (int idx = tid; idx < 64 * NBASIS; idx += 128) {
    int r = idx / NBASIS, n = idx % NBASIS;
    sR[r][n] = Rws[((size_t)b * TQ + rBlk + r) * 160 + h * NBASIS + n];
  }

#if __has_builtin(__builtin_amdgcn_tensor_load_to_lds)
  // ---- TDM: 1-D tensor copy b_nd[1280 x f32] -> sBnd (ISA ch. 7/8, D#) ----
  if (wave == 0) {
    // LDS aperture: flat addr low 32 bits are the LDS byte address.
    unsigned lds_off = (unsigned)(size_t)(&sBnd[0][0]);
    unsigned long long ga = (unsigned long long)(size_t)b_nd;
    u32x4 g0;
    g0[0] = 1u;                                   // count=1, user descriptor
    g0[1] = lds_off;                              // lds_addr
    g0[2] = (unsigned)(ga & 0xFFFFFFFFu);         // global_addr[31:0]
    g0[3] = (unsigned)((ga >> 32) & 0x01FFFFFFu)  // global_addr[56:32]
            | 0x80000000u;                        // type=2 ("image")
    const unsigned NEL = 1280u;                   // 10*128 elements
    i32x8 g1;
    g1[0] = (int)(2u << 16);                      // data_size=2 (4B), mask=0
    g1[1] = (int)((NEL & 0xFFFFu) << 16);         // tensor_dim0[15:0]
    g1[2] = (int)((NEL >> 16) | (1u << 16));      // tensor_dim0[31:16], tensor_dim1=1
    g1[3] = (int)(NEL << 16);                     // tile_dim0 = 1280
    g1[4] = 0;                                    // tile_dim1=0 (1-D), tile_dim2=0
    g1[5] = (int)NEL;                             // tensor_dim0_stride
    g1[6] = 0;
    g1[7] = 0;
    i32x4 gz = {0, 0, 0, 0};
#if __clang_major__ >= 23
    i32x8 gz8 = {0, 0, 0, 0, 0, 0, 0, 0};
    __builtin_amdgcn_tensor_load_to_lds(g0, g1, gz, gz, gz8, 0);
#else
    __builtin_amdgcn_tensor_load_to_lds(g0, g1, gz, gz, 0);
#endif
#if __has_builtin(__builtin_amdgcn_s_wait_tensorcnt)
    __builtin_amdgcn_s_wait_tensorcnt(0);
#endif
  }
#else
  for (int idx = tid; idx < NBASIS * MAXL; idx += 128)
    sBnd[idx / MAXL][idx % MAXL] = b_nd[idx];
#endif
  __syncthreads();

  // --- Q fragments for both K-steps (HD = 64 = 2 x 32) ---
  const int mA    = lane & 15;               // A-frag row
  const int koffA = (lane & 16) ? 8 : 0;     // A-frag K offset
  const int nB    = lane & 15;               // B-frag / C col
  const int kB0   = (lane & 16) ? 8 : 0;     // B-frag K offset
  const int mC    = (lane & 16) ? 8 : 0;     // C row offset

  v16bf aq[2];
  {
    const float* qrow = q + ((size_t)b * TQ + rW + mA) * E + h * HDIM;
#pragma unroll
    for (int kk = 0; kk < 2; ++kk) {
      v8u t;
#pragma unroll
      for (int v = 0; v < 8; ++v) {
        int k0 = ((v & 4) ? 16 : 0) + (v & 3) * 2 + koffA + kk * 32;
        float2 f = *(const float2*)(qrow + k0);
        t[v] = pack2(f.x, f.y);
      }
      aq[kk] = __builtin_bit_cast(v16bf, t);
    }
  }

  // Hoisted K_full/V_full bases: K_full[j] = j<128 ? k_cache[b,j] : k_new[b,j-128]
  const float* Kc = k_cache + (size_t)b * MAXL * E + h * HDIM;
  const float* Kn = k_new   + (size_t)b * TQ   * E + h * HDIM - (size_t)MAXL * E;
  const float* Vc = v_cache + (size_t)b * MAXL * E;
  const float* Vn = v_new   + (size_t)b * TQ   * E - (size_t)MAXL * E;

  // --- S = q @ K^T : 16 tiles x 2 WMMAs ---
  v8f S[16];
#pragma unroll
  for (int jt = 0; jt < 16; ++jt) {
    v8f c = kZero8;
    int j = jt * 16 + nB;
    const float* Krow = ((j < MAXL) ? Kc : Kn) + (size_t)j * E;
#pragma unroll
    for (int kk = 0; kk < 2; ++kk) {
      v8u t;
#pragma unroll
      for (int v = 0; v < 8; ++v) {
        float2 f = *(const float2*)(Krow + kB0 + ((v & 4) ? 16 : 0) + (v & 3) * 2 + kk * 32);
        t[v] = pack2(f.x, f.y);
      }
      v16bf bk = __builtin_bit_cast(v16bf, t);
      c = WMMA_BF16(aq[kk], bk, c);
    }
    S[jt] = c;
  }

  // --- scale + band mask + R.b_nd bias; row max ---
  float rmax[8], rsum[8], rinv[8];
#pragma unroll
  for (int v = 0; v < 8; ++v) rmax[v] = -3.0e38f;
#pragma unroll
  for (int jt = 0; jt < 16; ++jt) {
#pragma unroll
    for (int v = 0; v < 8; ++v) {
      int m = v + mC;                         // row within wave tile
      int i = rW + m;                         // q position
      int j = jt * 16 + nB;                   // kv position
      int d = MAXL + i - j;                   // band distance; band <=> d in [0,128)
      float val;
      if (d >= 0 && d < MAXL) {
        float ex = 0.f;
#pragma unroll
        for (int n = 0; n < NBASIS; ++n) ex += sR[wave * 16 + m][n] * sBnd[n][d];
        val = S[jt][v] * 0.125f + ex;         // 1/sqrt(64)
      } else {
        val = -1.0e9f;
      }
      S[jt][v] = val;
      rmax[v] = fmaxf(rmax[v], val);
    }
  }
#pragma unroll
  for (int v = 0; v < 8; ++v)
#pragma unroll
    for (int off = 1; off < 16; off <<= 1)
      rmax[v] = fmaxf(rmax[v], __shfl_xor(rmax[v], off, 32));

  // --- exp, row sums, spill P (bf16) to LDS in A-consumable layout ---
#pragma unroll
  for (int v = 0; v < 8; ++v) rsum[v] = 0.f;
#pragma unroll
  for (int jt = 0; jt < 16; ++jt) {
#pragma unroll
    for (int v = 0; v < 8; ++v) {
      float p = __expf(S[jt][v] - rmax[v]);
      rsum[v] += p;
      sP[wave][v + mC][jt * 16 + nB] = f2bf(p);
    }
  }
#pragma unroll
  for (int v = 0; v < 8; ++v) {
#pragma unroll
    for (int off = 1; off < 16; off <<= 1) rsum[v] += __shfl_xor(rsum[v], off, 32);
    rinv[v] = 1.0f / rsum[v];
  }
  // P produced & consumed by the same wave; LDS ops are in-order per wave.

  // --- A = P @ V : K = 256 (8 steps) x 4 HD tiles ---
  v8f acc2[4];
#pragma unroll
  for (int nt = 0; nt < 4; ++nt) acc2[nt] = kZero8;

  const unsigned int* Prow = (const unsigned int*)&sP[wave][mA][0];  // 128 pairs
#pragma unroll
  for (int kt = 0; kt < 8; ++kt) {
    int kp = (koffA >> 1) + kt * 16;          // 16B-aligned quad offsets
    v16bf ap = frag16(Prow, kp, kp + 8);
    const float* Vb = (kt < 4) ? Vc : Vn;     // uniform select (pairs never straddle 128)
#pragma unroll
    for (int nt = 0; nt < 4; ++nt) {
      int col = h * HDIM + nt * 16 + nB;
      v8u tb;
#pragma unroll
      for (int v = 0; v < 8; ++v) {
        int k = kt * 32 + kB0 + ((v & 4) ? 16 : 0) + (v & 3) * 2;        // T index
        const float* V0 = Vb + (size_t)k * E + col;
        tb[v] = pack2(V0[0], V0[E]);          // rows k, k+1
      }
      v16bf bv = __builtin_bit_cast(v16bf, tb);
      acc2[nt] = WMMA_BF16(ap, bv, acc2[nt]);
    }
  }

  // --- normalize by rowsum, write A ---
#pragma unroll
  for (int nt = 0; nt < 4; ++nt) {
#pragma unroll
    for (int v = 0; v < 8; ++v) {
      int row = rW + v + mC;
      int col = h * HDIM + nt * 16 + nB;
      Aout[((size_t)b * TQ + row) * E + col] = acc2[nt][v] * rinv[v];
    }
  }
}

__global__ void fill_ones(float* __restrict__ p, int n) {
  int i = blockIdx.x * blockDim.x + threadIdx.x;
  if (i < n) p[i] = 1.0f;
}

// ---------------------------------------------------------------------------
// Launch. Outputs concat: out[64,128,1024] | mask[64,1,128] | k_out | v_out.
// Since t == MAXLEN: k_out = k_new, v_out = v_new, mask = all ones.
// ---------------------------------------------------------------------------
extern "C" void kernel_launch(void* const* d_in, const int* in_sizes, int n_in,
                              void* d_out, int out_size, void* d_ws, size_t ws_size,
                              hipStream_t stream) {
  const float* input   = (const float*)d_in[0];
  const float* k_cache = (const float*)d_in[3];
  const float* v_cache = (const float*)d_in[4];
  const float* Wq      = (const float*)d_in[5];
  const float* bq      = (const float*)d_in[6];
  const float* Wk      = (const float*)d_in[7];
  const float* Wv      = (const float*)d_in[8];
  const float* Wproj   = (const float*)d_in[9];
  const float* bproj   = (const float*)d_in[10];
  const float* Wr      = (const float*)d_in[11];
  const float* br      = (const float*)d_in[12];
  const float* b_nd    = (const float*)d_in[13];

  const size_t BTE = (size_t)64 * 128 * 1024;  // 8,388,608
  float* out      = (float*)d_out;
  float* mask_out = out + BTE;
  float* k_out    = mask_out + 64 * 128;
  float* v_out    = k_out + BTE;

  float* q_ws = (float*)d_ws;                  // [B,128,1024]
  float* R_ws = q_ws + BTE;                    // [B,128,160]
  float* A_ws = R_ws + (size_t)64 * 128 * 160; // [B,128,1024]

  const int M = 8192, K = 1024;
  dim3 blk256(256), blk128(128);

  // Projections (k_new/v_new written straight into k_out/v_out).
  wmma_gemm_bf16<<<(M / 128) * (1024 / 128), blk256, 0, stream>>>(input, Wq, bq, nullptr, q_ws, M, 1024, K);
  wmma_gemm_bf16<<<(M / 128) * (1024 / 128), blk256, 0, stream>>>(input, Wk, nullptr, nullptr, k_out, M, 1024, K);
  wmma_gemm_bf16<<<(M / 128) * (1024 / 128), blk256, 0, stream>>>(input, Wv, nullptr, nullptr, v_out, M, 1024, K);
  wmma_gemm_bf16<<<(M / 128) * 2, blk256, 0, stream>>>(input, Wr, br, nullptr, R_ws, M, 160, K);

  // Attention core: B*H*2 blocks of 128 threads (4 waves x 16 q-rows).
  wmma_attention<<<64 * 16 * 2, blk128, 0, stream>>>(q_ws, k_cache, v_cache, k_out, v_out, R_ws, b_nd, A_ws);

  // Output projection + residual.
  wmma_gemm_bf16<<<(M / 128) * (1024 / 128), blk256, 0, stream>>>(A_ws, Wproj, bproj, input, out, M, 1024, K);

  // new_state_mask: all true (t == MAXLEN drops all old state, appends ones).
  fill_ones<<<(64 * 128 + 255) / 256, blk256, 0, stream>>>(mask_out, 64 * 128);
}
